// MoE_592705487380
// MI455X (gfx1250) — matrix operand order repaired and checked by reading, not compiled, not run on hardware
//
#include <hip/hip_runtime.h>
#include <hip/hip_bf16.h>

// ---------------- problem constants ----------------
#define NTOK 4096      // B*S
#define DIM  1024      // D
#define NEXP 16        // E
#define FF   512       // F
#define TOPK 4
#define MT   32        // tokens per FFN block (2 row-tiles -> 2x B reuse)
#define SLOT_CAP (NTOK * TOPK + MT * NEXP)   // 16896 (pad each expert to 32-row tiles)

typedef __attribute__((ext_vector_type(16))) __bf16 v16bf;
typedef __attribute__((ext_vector_type(8)))  __bf16 v8bf;
typedef __attribute__((ext_vector_type(8)))  float  v8f;
typedef __attribute__((ext_vector_type(4)))  int    v4i;

#if defined(__has_builtin)
#if __has_builtin(__builtin_amdgcn_global_load_async_to_lds_b128) && \
    __has_builtin(__builtin_amdgcn_s_wait_asynccnt)
#define USE_ASYNC_LDS 1
#endif
#endif

typedef __attribute__((address_space(1))) v4i* as1_v4i_ptr;
typedef __attribute__((address_space(3))) v4i* as3_v4i_ptr;

static __device__ __forceinline__ v16bf cat16(v8bf lo, v8bf hi) {
    return __builtin_shufflevector(lo, hi, 0,1,2,3,4,5,6,7,8,9,10,11,12,13,14,15);
}

static __device__ __forceinline__ v8f wmma_bf16(v16bf a, v16bf b, v8f c) {
    return __builtin_amdgcn_wmma_f32_16x16x32_bf16(false, a, false, b, (short)0, c,
                                                   false, false);
}

// ---------------- 1) x fp32 -> bf16 ----------------
__global__ __launch_bounds__(256) void k_cvtx(const float* __restrict__ x,
                                              __bf16* __restrict__ xb) {
    int i = (blockIdx.x * 256 + threadIdx.x) * 4;
    float4 v = *reinterpret_cast<const float4*>(x + i);
    xb[i + 0] = (__bf16)v.x;
    xb[i + 1] = (__bf16)v.y;
    xb[i + 2] = (__bf16)v.z;
    xb[i + 3] = (__bf16)v.w;
}

// ---------------- 2) per-expert transpose + convert: src [R,C] f32 -> dst [C,R] bf16
__global__ __launch_bounds__(256) void k_tconv(const float* __restrict__ src,
                                               __bf16* __restrict__ dst,
                                               int R, int C) {
    __shared__ float tile[32][33];
    const int e = blockIdx.z;
    const float* s = src + (size_t)e * R * C;
    __bf16* d = dst + (size_t)e * R * C;
    const int c0 = blockIdx.x * 32, r0 = blockIdx.y * 32;
    const int tx = threadIdx.x, ty = threadIdx.y;  // 32 x 8
#pragma unroll
    for (int i = 0; i < 32; i += 8)
        tile[ty + i][tx] = s[(size_t)(r0 + ty + i) * C + (c0 + tx)];
    __syncthreads();
#pragma unroll
    for (int i = 0; i < 32; i += 8)
        d[(size_t)(c0 + ty + i) * R + (r0 + tx)] = (__bf16)tile[tx][ty + i];
}

// ---------------- 3) router: logits -> softmax -> top4 -> renorm ----------------
__global__ __launch_bounds__(32) void k_router(const float* __restrict__ x,
                                               const float* __restrict__ Wr,
                                               int* __restrict__ topk_idx,
                                               float* __restrict__ topk_w) {
    const int t = blockIdx.x;
    const int lane = threadIdx.x;
    __shared__ float sl[NEXP];
    if (lane < NEXP) {
        const float* xr = x + (size_t)t * DIM;
        float acc = 0.f;
        for (int d = 0; d < DIM; ++d) acc += xr[d] * Wr[d * NEXP + lane];
        sl[lane] = acc;
    }
    __syncthreads();
    if (lane == 0) {
        float m = sl[0];
        for (int i = 1; i < NEXP; ++i) m = fmaxf(m, sl[i]);
        float p[NEXP];
        for (int i = 0; i < NEXP; ++i) p[i] = __expf(sl[i] - m);
        // top-4 by value, first-index tie-break (matches jax top_k); softmax
        // denominator cancels in the renormalization.
        bool taken[NEXP] = {};
        int sel[TOPK]; float sw[TOPK]; float wsum = 0.f;
        for (int k = 0; k < TOPK; ++k) {
            int best = 0; float bv = -1.f;
            for (int i = 0; i < NEXP; ++i)
                if (!taken[i] && p[i] > bv) { bv = p[i]; best = i; }
            taken[best] = true; sel[k] = best; sw[k] = bv; wsum += bv;
        }
        for (int k = 0; k < TOPK; ++k) {
            topk_idx[t * TOPK + k] = sel[k];
            topk_w[t * TOPK + k] = sw[k] / wsum;
        }
    }
}

// ---------------- 4) per-expert token count (deterministic) ----------------
__global__ __launch_bounds__(256) void k_count(const int* __restrict__ topk_idx,
                                               int* __restrict__ counts) {
    const int e = blockIdx.x;
    __shared__ int red[256];
    int c = 0;
    for (int t = threadIdx.x; t < NTOK; t += 256) {
        const int* q = topk_idx + t * TOPK;
        if (q[0] == e || q[1] == e || q[2] == e || q[3] == e) ++c;
    }
    red[threadIdx.x] = c;
    __syncthreads();
    for (int s = 128; s > 0; s >>= 1) {
        if (threadIdx.x < s) red[threadIdx.x] += red[threadIdx.x + s];
        __syncthreads();
    }
    if (threadIdx.x == 0) counts[e] = red[0];
}

// ---------------- 5) tiny scan: base offsets padded to MT-token tiles ----------------
__global__ void k_scan(const int* __restrict__ counts,
                       int* __restrict__ base, int* __restrict__ tiles) {
    int total = 0;
    for (int e = 0; e < NEXP; ++e) {
        base[e] = total;
        tiles[e] = (counts[e] + MT - 1) / MT;
        total += tiles[e] * MT;
    }
}

// ---------------- 6) deterministic compaction (token order, ballot/popc) -------------
__global__ __launch_bounds__(32) void k_fill(const int* __restrict__ topk_idx,
                                             const float* __restrict__ topk_w,
                                             const int* __restrict__ base,
                                             const int* __restrict__ tiles,
                                             int* __restrict__ idx_slot,
                                             float* __restrict__ cw_slot,
                                             int* __restrict__ tok2slot) {
    const int e = blockIdx.x;
    const int lane = threadIdx.x;
    int pos = base[e];
    for (int t0 = 0; t0 < NTOK; t0 += 32) {
        const int t = t0 + lane;
        const int* q = topk_idx + t * TOPK;
        int kpos = -1;
#pragma unroll
        for (int k = 0; k < TOPK; ++k)
            if (q[k] == e) kpos = k;
        const bool m = (kpos >= 0);
        const unsigned mask = (unsigned)__ballot(m);
        if (m) {
            const int slot = pos + __popc(mask & ((1u << lane) - 1u));
            idx_slot[slot] = t;
            cw_slot[slot] = topk_w[t * TOPK + kpos];
            tok2slot[t * TOPK + kpos] = slot;
        }
        pos += __popc(mask);
    }
    const int pend = base[e] + tiles[e] * MT;
    for (int s = pos + lane; s < pend; s += 32) {  // pad rows: token 0, weight 0
        idx_slot[s] = 0;
        cw_slot[s] = 0.f;
    }
}

// ---------------- 7) fused expert FFN: silu(x Wg) * (x Wu) then (.) Wd * c -----------
// grid = (max_tiles, E), block = 256 (8 waves). 32-token tile per block.
#define AP 1032   // A-panel LDS stride (halves): 2064 B (16B aligned, bank-spread)
#define HP 520    // H LDS stride (halves): 1040 B
__global__ __launch_bounds__(256) void k_ffn(const __bf16* __restrict__ xb,
                                             const __bf16* __restrict__ WgT,
                                             const __bf16* __restrict__ WuT,
                                             const __bf16* __restrict__ WdT,
                                             const int* __restrict__ idx_slot,
                                             const float* __restrict__ cw_slot,
                                             const int* __restrict__ base,
                                             const int* __restrict__ tiles,
                                             __bf16* __restrict__ contrib) {
    const int e = blockIdx.y;
    const int mt = blockIdx.x;
    if (mt >= tiles[e]) return;
    const int slot0 = base[e] + mt * MT;

    __shared__ __align__(16) __bf16 As[MT][AP];  // 32 x 1024 gathered x rows (bf16)
    __shared__ __align__(16) __bf16 Hs[MT][HP];  // 32 x 512 hidden (bf16)
    __shared__ float cs[MT];
    __shared__ int ts[MT];

    const int tid = threadIdx.x;
    if (tid < MT) {
        ts[tid] = idx_slot[slot0 + tid];
        cs[tid] = cw_slot[slot0 + tid];
    }
    __syncthreads();

    // cooperative load of A panel: MT rows x 1024 halves, 16-byte chunks
#if USE_ASYNC_LDS
    for (int c = tid; c < MT * 128; c += 256) {
        const int r = c >> 7;
        const int koff = (c & 127) * 8;
        v4i* gp = (v4i*)(xb + (size_t)ts[r] * DIM + koff);  // const cast via C-style
        v4i* lp = (v4i*)&As[r][koff];
        __builtin_amdgcn_global_load_async_to_lds_b128((as1_v4i_ptr)gp, (as3_v4i_ptr)lp,
                                                       0, 0);
    }
    __builtin_amdgcn_s_wait_asynccnt(0);
#else
    for (int c = tid; c < MT * 128; c += 256) {
        const int r = c >> 7;
        const int koff = (c & 127) * 8;
        *reinterpret_cast<v8bf*>(&As[r][koff]) =
            *reinterpret_cast<const v8bf*>(xb + (size_t)ts[r] * DIM + koff);
    }
#endif
    __syncthreads();

    const int wave = tid >> 5;
    const int lane = tid & 31;
    const int lrow = lane & 15;   // A row / B col / C col within 16-tile
    const int kg = lane >> 4;     // K-half: k = 16*kg + s (contiguous, same for A and B)

    // ---- phase A: H[32,512] = silu(A Wg) * (A Wu) ----
    // wave covers 32 cols (2 col-tiles) x 32 rows (2 row-tiles); N-loop x2.
    {
        const __bf16* wg = WgT + (size_t)e * FF * DIM;
        const __bf16* wu = WuT + (size_t)e * FF * DIM;
        for (int nl = 0; nl < 2; ++nl) {
            const int nb0 = nl * 256 + wave * 32;
            v8f accg[2][2] = {};
            v8f accu[2][2] = {};
            v16bf bg0[2], bu0[2], bg1[2], bu1[2];  // double-buffered B frags
            auto loadB = [&](int k0, v16bf (&bg)[2], v16bf (&bu)[2]) {
#pragma unroll
                for (int n = 0; n < 2; ++n) {
                    const __bf16* pg = wg + (size_t)(nb0 + n * 16 + lrow) * DIM + k0 + kg * 16;
                    const __bf16* pu = wu + (size_t)(nb0 + n * 16 + lrow) * DIM + k0 + kg * 16;
                    bg[n] = cat16(*(const v8bf*)pg, *(const v8bf*)(pg + 8));
                    bu[n] = cat16(*(const v8bf*)pu, *(const v8bf*)(pu + 8));
                }
            };
            auto step = [&](int k0, v16bf (&bg)[2], v16bf (&bu)[2]) {
                v16bf a[2];
#pragma unroll
                for (int m = 0; m < 2; ++m)
                    a[m] = cat16(*(const v8bf*)&As[m * 16 + lrow][k0 + kg * 16],
                                 *(const v8bf*)&As[m * 16 + lrow][k0 + kg * 16 + 8]);
#pragma unroll
                for (int m = 0; m < 2; ++m)
#pragma unroll
                    for (int n = 0; n < 2; ++n) {
                        accg[m][n] = wmma_bf16(a[m], bg[n], accg[m][n]);
                        accu[m][n] = wmma_bf16(a[m], bu[n], accu[m][n]);
                    }
            };
            loadB(0, bg0, bu0);
            for (int kt = 0; kt < 32; kt += 2) {
                const int k0 = kt * 32;
                loadB(k0 + 32, bg1, bu1);        // prefetch odd step
                step(k0, bg0, bu0);
                if (kt + 2 < 32) loadB(k0 + 64, bg0, bu0);  // prefetch next even
                step(k0 + 32, bg1, bu1);
            }
#pragma unroll
            for (int m = 0; m < 2; ++m)
#pragma unroll
                for (int n = 0; n < 2; ++n)
#pragma unroll
                    for (int r = 0; r < 8; ++r) {
                        const float g = accg[m][n][r];
                        const float u = accu[m][n][r];
                        const float h = (g / (1.0f + __expf(-g))) * u;  // silu(g)*u
                        Hs[m * 16 + r + 8 * kg][nb0 + n * 16 + lrow] = (__bf16)h;
                    }
        }
    }
    __syncthreads();

    // ---- phase B: contrib[slot, 0:1024] = c * (H Wd) ----
    // wave covers 64 cols (4 col-tiles) x 32 rows (2 row-tiles); N-loop x2.
    {
        const __bf16* wd = WdT + (size_t)e * DIM * FF;
        for (int nl = 0; nl < 2; ++nl) {
            const int db0 = nl * 512 + wave * 64;
            v8f acc[2][4] = {};
            v16bf bd0[4], bd1[4];  // double-buffered B frags
            auto loadB = [&](int k0, v16bf (&bd)[4]) {
#pragma unroll
                for (int n = 0; n < 4; ++n) {
                    const __bf16* bp = wd + (size_t)(db0 + n * 16 + lrow) * FF + k0 + kg * 16;
                    bd[n] = cat16(*(const v8bf*)bp, *(const v8bf*)(bp + 8));
                }
            };
            auto step = [&](int k0, v16bf (&bd)[4]) {
                v16bf a[2];
#pragma unroll
                for (int m = 0; m < 2; ++m)
                    a[m] = cat16(*(const v8bf*)&Hs[m * 16 + lrow][k0 + kg * 16],
                                 *(const v8bf*)&Hs[m * 16 + lrow][k0 + kg * 16 + 8]);
#pragma unroll
                for (int m = 0; m < 2; ++m)
#pragma unroll
                    for (int n = 0; n < 4; ++n)
                        acc[m][n] = wmma_bf16(a[m], bd[n], acc[m][n]);
            };
            loadB(0, bd0);
            for (int kt = 0; kt < 16; kt += 2) {
                const int k0 = kt * 32;
                loadB(k0 + 32, bd1);
                step(k0, bd0);
                if (kt + 2 < 16) loadB(k0 + 64, bd0);
                step(k0 + 32, bd1);
            }
#pragma unroll
            for (int m = 0; m < 2; ++m)
#pragma unroll
                for (int n = 0; n < 4; ++n)
#pragma unroll
                    for (int r = 0; r < 8; ++r) {
                        const int row = m * 16 + r + 8 * kg;
                        const float v = acc[m][n][r] * cs[row];
                        contrib[(size_t)(slot0 + row) * DIM + db0 + n * 16 + lrow] =
                            (__bf16)v;
                    }
        }
    }
}

// ---------------- 8) combine: out[t,d] = sum_k contrib[slot(t,k), d] ----------------
__global__ __launch_bounds__(256) void k_combine(const __bf16* __restrict__ contrib,
                                                 const int* __restrict__ tok2slot,
                                                 float* __restrict__ out) {
    const int gid = blockIdx.x * 256 + threadIdx.x;  // < NTOK*DIM
    const int t = gid >> 10;
    const int d = gid & (DIM - 1);
    const int* s = tok2slot + t * TOPK;
    float acc = 0.f;
#pragma unroll
    for (int k = 0; k < TOPK; ++k)
        acc += (float)contrib[(size_t)s[k] * DIM + d];
    out[gid] = acc;
}

// ---------------- launch ----------------
extern "C" void kernel_launch(void* const* d_in, const int* in_sizes, int n_in,
                              void* d_out, int out_size, void* d_ws, size_t ws_size,
                              hipStream_t stream) {
    const float* x  = (const float*)d_in[0];  // [B,S,D]
    const float* Wr = (const float*)d_in[1];  // [D,E]
    const float* Wg = (const float*)d_in[2];  // [E,D,F]
    const float* Wu = (const float*)d_in[3];  // [E,D,F]
    const float* Wd = (const float*)d_in[4];  // [E,F,D]
    float* out = (float*)d_out;

    // workspace carve-up (256-byte aligned)
    char* ws = (char*)d_ws;
    size_t off = 0;
    auto carve = [&](size_t bytes) -> char* {
        char* p = ws + off;
        off += (bytes + 255) & ~(size_t)255;
        return p;
    };
    __bf16* xb  = (__bf16*)carve((size_t)NTOK * DIM * 2);
    __bf16* WgT = (__bf16*)carve((size_t)NEXP * FF * DIM * 2);   // [E,F,D]
    __bf16* WuT = (__bf16*)carve((size_t)NEXP * FF * DIM * 2);   // [E,F,D]
    __bf16* WdT = (__bf16*)carve((size_t)NEXP * DIM * FF * 2);   // [E,D,F]
    int*   topk_idx = (int*)carve((size_t)NTOK * TOPK * 4);
    float* topk_w   = (float*)carve((size_t)NTOK * TOPK * 4);
    int*   counts   = (int*)carve(NEXP * 4);
    int*   basev    = (int*)carve(NEXP * 4);
    int*   tilesv   = (int*)carve(NEXP * 4);
    int*   idx_slot = (int*)carve((size_t)SLOT_CAP * 4);
    float* cw_slot  = (float*)carve((size_t)SLOT_CAP * 4);
    int*   tok2slot = (int*)carve((size_t)NTOK * TOPK * 4);
    __bf16* contrib = (__bf16*)carve((size_t)SLOT_CAP * DIM * 2);
    (void)ws_size; (void)in_sizes; (void)n_in; (void)out_size;

    // 1) x -> bf16
    k_cvtx<<<(NTOK * DIM) / (256 * 4), 256, 0, stream>>>(x, xb);
    // 2) weight transpose+convert
    k_tconv<<<dim3(FF / 32, DIM / 32, NEXP), dim3(32, 8), 0, stream>>>(Wg, WgT, DIM, FF);
    k_tconv<<<dim3(FF / 32, DIM / 32, NEXP), dim3(32, 8), 0, stream>>>(Wu, WuT, DIM, FF);
    k_tconv<<<dim3(DIM / 32, FF / 32, NEXP), dim3(32, 8), 0, stream>>>(Wd, WdT, FF, DIM);
    // 3) router
    k_router<<<NTOK, 32, 0, stream>>>(x, Wr, topk_idx, topk_w);
    // 4-6) deterministic expert binning
    k_count<<<NEXP, 256, 0, stream>>>(topk_idx, counts);
    k_scan<<<1, 1, 0, stream>>>(counts, basev, tilesv);
    k_fill<<<NEXP, 32, 0, stream>>>(topk_idx, topk_w, basev, tilesv,
                                    idx_slot, cw_slot, tok2slot);
    // 7) fused SwiGLU expert FFN (WMMA)
    k_ffn<<<dim3(NTOK / MT, NEXP), 256, 0, stream>>>(xb, WgT, WuT, WdT,
                                                     idx_slot, cw_slot,
                                                     basev, tilesv, contrib);
    // 8) combine top-k contributions per token
    k_combine<<<(NTOK * DIM) / 256, 256, 0, stream>>>(contrib, tok2slot, out);
}
